// MVTS_GCN_RNN_80616536146448
// MI455X (gfx1250) — matrix operand order repaired and checked by reading, not compiled, not run on hardware
//
#include <hip/hip_runtime.h>
#include <hip/hip_bf16.h>

// ---------------------------------------------------------------------------
// MVTS GCN-RNN for MI455X (gfx1250, wave32, WMMA).
//  - bf16 WMMA (f32 accumulate) for all large GEMMs, operands pre-converted
//    to bf16 (B pre-transposed to column-major) so LDS staging and fragment
//    loads are all contiguous 16-byte b128 transactions
//  - GLOBAL_LOAD_ASYNC_TO_LDS_B128 staging (builtin confirmed present;
//    signature: (v4i AS1*, v4i AS3*, i32, i32))
//  - persistent grid-synced kernel for the 8192-step LSTM recurrence
//  - atomic scatter for GCN edge aggregation
// ---------------------------------------------------------------------------

typedef __attribute__((ext_vector_type(16))) __bf16 v16bf;
typedef __attribute__((ext_vector_type(8)))  __bf16 v8bf;
typedef __attribute__((ext_vector_type(8)))  float  v8f;
typedef __attribute__((__vector_size__(16))) int    v4i_t;

#define NUM_NODES 1024
#define SEQ_LEN   8192
#define SEQ_EMB   1024
#define GCN_H     1024
#define EVENT_EMB 1024
#define NUM_CLASSES 16
#define NUM_EDGES 16384
#define GATES     (4 * SEQ_EMB)      // 4096

// ---- gfx1250 async global->LDS staging (probe-guarded) ----
#if defined(__has_builtin)
#  if __has_builtin(__builtin_amdgcn_global_load_async_to_lds_b128)
#    define USE_ASYNC_LDS 1
#  endif
#endif
#ifndef USE_ASYNC_LDS
#  define USE_ASYNC_LDS 0
#endif

#if USE_ASYNC_LDS
__device__ __forceinline__ void async_copy_b128(const void* g, void* l) {
    __builtin_amdgcn_global_load_async_to_lds_b128(
        (__attribute__((address_space(1))) v4i_t*)(g),
        (__attribute__((address_space(3))) v4i_t*)(l), 0, 0);
}
__device__ __forceinline__ void wait_async_all() {
#if __has_builtin(__builtin_amdgcn_s_wait_asynccnt)
    __builtin_amdgcn_s_wait_asynccnt(0);
#else
    asm volatile("s_wait_asynccnt 0x0" ::: "memory");
#endif
}
#endif

// ------------------------------- utilities --------------------------------

__global__ void k_zero_u32(unsigned* __restrict__ p, long long n) {
    long long i = (long long)blockIdx.x * blockDim.x + threadIdx.x;
    if (i < n) p[i] = 0u;
}

__global__ void k_bias_sum(const float* __restrict__ a, const float* __restrict__ b,
                           float* __restrict__ out, int n) {
    int i = blockIdx.x * blockDim.x + threadIdx.x;
    if (i < n) out[i] = a[i] + b[i];
}

__global__ void k_deg_init(float* __restrict__ deg, int n) {
    int i = blockIdx.x * blockDim.x + threadIdx.x;
    if (i < n) deg[i] = 1.0f;              // self loop
}
__global__ void k_deg_scatter(const int* __restrict__ tgt, float* __restrict__ deg, int e) {
    int i = blockIdx.x * blockDim.x + threadIdx.x;
    if (i < e) atomicAdd(&deg[tgt[i]], 1.0f);
}
__global__ void k_rsqrt(const float* __restrict__ deg, float* __restrict__ dinv, int n) {
    int i = blockIdx.x * blockDim.x + threadIdx.x;
    if (i < n) dinv[i] = rsqrtf(deg[i]);
}

// f32 -> bf16 convert (contiguous)
__global__ void k_convert_bf16(const float* __restrict__ in, __bf16* __restrict__ out,
                               long long n) {
    long long i = (long long)blockIdx.x * blockDim.x + threadIdx.x;
    if (i < n) out[i] = (__bf16)in[i];
}

// in[R][Cc] f32 -> out[Cc][R] bf16, LDS-tiled 32x32, 256 threads (32x8)
__global__ __launch_bounds__(256) void k_transpose_bf16(
    const float* __restrict__ in, __bf16* __restrict__ out, int R, int Cc) {
    __shared__ float tile[32][33];
    const int c0 = blockIdx.x * 32, r0 = blockIdx.y * 32;
    const int tx = threadIdx.x & 31, ty = threadIdx.x >> 5;
    for (int i = ty; i < 32; i += 8)
        tile[i][tx] = in[(long long)(r0 + i) * Cc + c0 + tx];
    __syncthreads();
    for (int i = ty; i < 32; i += 8)
        out[(long long)(c0 + i) * R + r0 + tx] = (__bf16)tile[tx][i];
}

// ------------------------- bf16 WMMA GEMM ---------------------------------
// C[M,N] f32 = A @ B (+bias[n]).  A: bf16 [M][K] row-major (lda).
// Bt: bf16 [N][K] (B column-major, ldbt) -> staging needs no transpose and
// both A and B fragments become contiguous K-runs (this code's k = 16*half + s, same in A and B).
// Block: 256 thr = 8 waves (2x4), block tile 64x128, wave tile 32x32,
// K-chunk 64 -> 8 v_wmma per wave per chunk.  M%64==0, N%128==0, K%64==0.

#define BM 64
#define BN 128
#define KC 64
#define LDK (KC + 8)     // 72 elems = 144 B rows: 16B-aligned, bank-spread

__global__ __launch_bounds__(256) void k_gemm_bf16_wmma(
    const __bf16* __restrict__ A, int lda,
    const __bf16* __restrict__ Bt, int ldbt,
    float* __restrict__ C, const float* __restrict__ bias,
    int M, int N, int K)
{
    __shared__ __align__(16) __bf16 As[BM][LDK];
    __shared__ __align__(16) __bf16 Bs[BN][LDK];   // Bs[n][k]

    const int tid  = threadIdx.x;
    const int lane = tid & 31;
    const int wid  = tid >> 5;
    const int wm   = wid >> 2;           // 0..1
    const int wn   = wid & 3;            // 0..3
    const int tile_m = blockIdx.x * BM;
    const int tile_n = blockIdx.y * BN;
    const int half = lane >> 4;
    const int l15  = lane & 15;

    v8f acc[2][2] = {};

    for (int k0 = 0; k0 < K; k0 += KC) {
        // ---- stage A (64x64) and Bt (128x64) as 16-byte chunks ----
#if USE_ASYNC_LDS
        #pragma unroll
        for (int q = tid; q < (BM * KC) / 8; q += 256) {        // 2 iters
            int r = q >> 3, c8 = (q & 7) * 8;
            async_copy_b128(A + (long long)(tile_m + r) * lda + k0 + c8, &As[r][c8]);
        }
        #pragma unroll
        for (int q = tid; q < (BN * KC) / 8; q += 256) {        // 4 iters
            int r = q >> 3, c8 = (q & 7) * 8;
            async_copy_b128(Bt + (long long)(tile_n + r) * ldbt + k0 + c8, &Bs[r][c8]);
        }
        wait_async_all();
#else
        #pragma unroll
        for (int q = tid; q < (BM * KC) / 8; q += 256) {
            int r = q >> 3, c8 = (q & 7) * 8;
            *(v8bf*)&As[r][c8] = *(const v8bf*)(A + (long long)(tile_m + r) * lda + k0 + c8);
        }
        #pragma unroll
        for (int q = tid; q < (BN * KC) / 8; q += 256) {
            int r = q >> 3, c8 = (q & 7) * 8;
            *(v8bf*)&Bs[r][c8] = *(const v8bf*)(Bt + (long long)(tile_n + r) * ldbt + k0 + c8);
        }
#endif
        __syncthreads();

        #pragma unroll
        for (int kk = 0; kk < KC; kk += 32) {
            // B fragments: one contiguous 16-elem K-run per lane
            v16bf bfrag[2];
            #pragma unroll
            for (int nt = 0; nt < 2; ++nt) {
                const int n = wn * 32 + nt * 16 + l15;
                v8bf lo = *(const v8bf*)&Bs[n][kk + 16 * half];
                v8bf hi = *(const v8bf*)&Bs[n][kk + 16 * half + 8];
                bfrag[nt] = __builtin_shufflevector(lo, hi,
                    0,1,2,3,4,5,6,7,8,9,10,11,12,13,14,15);
            }
            // A fragments: one contiguous 16-elem K-run per lane
            v16bf afrag[2];
            #pragma unroll
            for (int mt = 0; mt < 2; ++mt) {
                const int m = wm * 32 + mt * 16 + l15;
                v8bf lo = *(const v8bf*)&As[m][kk + 16 * half];
                v8bf hi = *(const v8bf*)&As[m][kk + 16 * half + 8];
                afrag[mt] = __builtin_shufflevector(lo, hi,
                    0,1,2,3,4,5,6,7,8,9,10,11,12,13,14,15);
            }
            #pragma unroll
            for (int mt = 0; mt < 2; ++mt)
                #pragma unroll
                for (int nt = 0; nt < 2; ++nt)
                    acc[mt][nt] = __builtin_amdgcn_wmma_f32_16x16x32_bf16(
                        false, afrag[mt], false, bfrag[nt], (short)0,
                        acc[mt][nt], false, false);
        }
        __syncthreads();
    }

    // ---- epilogue: D layout = lane(N), half*8+r (M) ----
    #pragma unroll
    for (int mt = 0; mt < 2; ++mt) {
        const int mbase = tile_m + wm * 32 + mt * 16 + half * 8;
        #pragma unroll
        for (int nt = 0; nt < 2; ++nt) {
            const int n = tile_n + wn * 32 + nt * 16 + l15;
            const float bv = bias ? bias[n] : 0.0f;
            #pragma unroll
            for (int r = 0; r < 8; ++r)
                C[(long long)(mbase + r) * N + n] = acc[mt][nt][r] + bv;
        }
    }
}

// ----------------------------- GCN aggregation -----------------------------

__global__ void k_gcn_scatter(const float* __restrict__ H, float* __restrict__ AGG,
                              const int* __restrict__ src, const int* __restrict__ tgt,
                              const float* __restrict__ dinv) {
    const int e = blockIdx.x;
    const int s = src[e], t = tgt[e];
    const float w = dinv[s] * dinv[t];
    const float* hr = H + (long long)s * GCN_H;
    float* ar = AGG + (long long)t * GCN_H;
    for (int f = threadIdx.x; f < GCN_H; f += blockDim.x)
        atomicAdd(&ar[f], w * hr[f]);
}

// out = AGG + dinv^2 * H + bias (+relu). outF (f32) and/or outB (bf16).
__global__ void k_gcn_finalize(const float* __restrict__ AGG, const float* __restrict__ H,
                               const float* __restrict__ dinv, const float* __restrict__ bias,
                               float* __restrict__ outF, __bf16* __restrict__ outB, int relu) {
    const int i = blockIdx.x;
    const float w = dinv[i] * dinv[i];
    for (int f = threadIdx.x; f < GCN_H; f += blockDim.x) {
        float v = AGG[(long long)i * GCN_H + f] + w * H[(long long)i * GCN_H + f] + bias[f];
        if (relu) v = fmaxf(v, 0.0f);
        if (outF) outF[(long long)i * GCN_H + f] = v;
        if (outB) outB[(long long)i * GCN_H + f] = (__bf16)v;
    }
}

// ------------------------ persistent LSTM recurrence -----------------------

#define LSTM_WGS 16   // co-resident workgroups; each owns 64 hidden units

__device__ __forceinline__ float sigf(float x) { return 1.0f / (1.0f + __expf(-x)); }

__device__ __forceinline__ void grid_barrier(unsigned* bar) {
    __syncthreads();
    if (threadIdx.x == 0) {
        __threadfence();
        unsigned gen = __atomic_load_n(&bar[1], __ATOMIC_ACQUIRE);
        unsigned a = __atomic_add_fetch(&bar[0], 1u, __ATOMIC_ACQ_REL);
        if (a == (unsigned)LSTM_WGS) {
            __atomic_store_n(&bar[0], 0u, __ATOMIC_RELAXED);
            __atomic_add_fetch(&bar[1], 1u, __ATOMIC_RELEASE);
        } else {
            while (__atomic_load_n(&bar[1], __ATOMIC_ACQUIRE) == gen)
                __builtin_amdgcn_s_sleep(2);
        }
    }
    __syncthreads();
}

__global__ __launch_bounds__(256) void k_lstm(
    const float* __restrict__ Whh,    // [4096, 1024]
    const float* __restrict__ G0,     // [T, 4096] = x_t@W_ih^T + (b_ih+b_hh)
    float* __restrict__ hbuf,         // [2][1024] double-buffered h (zeroed)
    unsigned* __restrict__ bar,       // [2]
    int T)
{
    __shared__ float hs[SEQ_EMB];
    __shared__ float gv[64][4];
    __shared__ float cs[64];

    const int tid     = threadIdx.x;
    const int u_local = tid >> 2;
    const int gate    = tid & 3;           // torch order i,f,g,o
    const int u       = blockIdx.x * 64 + u_local;
    const int row     = gate * SEQ_EMB + u;
    const float4* wrow = (const float4*)(Whh + (long long)row * SEQ_EMB);

    if (tid < 64) cs[tid] = 0.0f;

    for (int t = 0; t < T; ++t) {
        const volatile float* hin = hbuf + (t & 1) * SEQ_EMB;
        for (int i = tid; i < SEQ_EMB; i += 256) hs[i] = hin[i];
        __syncthreads();

        const float4* h4 = (const float4*)hs;
        float s = 0.0f;
        #pragma unroll 8
        for (int k = 0; k < SEQ_EMB / 4; ++k) {
            float4 w = wrow[k], h = h4[k];
            s = fmaf(w.x, h.x, s); s = fmaf(w.y, h.y, s);
            s = fmaf(w.z, h.z, s); s = fmaf(w.w, h.w, s);
        }
        gv[u_local][gate] = s + G0[(long long)t * GATES + row];
        __syncthreads();

        if (tid < 64) {
            const float gi = gv[tid][0], gf = gv[tid][1];
            const float gg = gv[tid][2], go = gv[tid][3];
            float c = cs[tid];
            c = sigf(gf) * c + sigf(gi) * tanhf(gg);
            const float h = sigf(go) * tanhf(c);
            cs[tid] = c;
            hbuf[((t + 1) & 1) * SEQ_EMB + blockIdx.x * 64 + tid] = h;
        }
        grid_barrier(bar);   // h_last ends in buffer 0 (T even)
    }
}

// ------------------------------ readout head -------------------------------

__global__ void k_col_mean(const float* __restrict__ X, float* __restrict__ out) {
    const int f = blockIdx.x * blockDim.x + threadIdx.x;
    if (f >= GCN_H) return;
    float s = 0.0f;
    for (int i = 0; i < NUM_NODES; ++i) s += X[(long long)i * GCN_H + f];
    out[f] = s * (1.0f / NUM_NODES);
}

__global__ void k_lin1_relu(const float* __restrict__ h_last, const float* __restrict__ gvec,
                            const float* __restrict__ W, const float* __restrict__ b,
                            float* __restrict__ ev) {
    const int o = blockIdx.x * blockDim.x + threadIdx.x;
    if (o >= EVENT_EMB) return;
    const float* wr = W + (long long)o * (SEQ_EMB + GCN_H);
    float s = b[o];
    #pragma unroll 4
    for (int i = 0; i < SEQ_EMB; ++i) s = fmaf(wr[i], h_last[i], s);
    #pragma unroll 4
    for (int i = 0; i < GCN_H; ++i)  s = fmaf(wr[SEQ_EMB + i], gvec[i], s);
    ev[o] = fmaxf(s, 0.0f);
}

__global__ void k_lin2_logsoftmax(const float* __restrict__ ev, const float* __restrict__ W,
                                  const float* __restrict__ b, float* __restrict__ out) {
    __shared__ float cls[NUM_CLASSES];
    const int o = threadIdx.x;
    if (o < NUM_CLASSES) {
        const float* wr = W + (long long)o * EVENT_EMB;
        float s = b[o];
        for (int i = 0; i < EVENT_EMB; ++i) s = fmaf(wr[i], ev[i], s);
        cls[o] = s;
    }
    __syncthreads();
    if (o == 0) {
        float mx = cls[0];
        for (int i = 1; i < NUM_CLASSES; ++i) mx = fmaxf(mx, cls[i]);
        float se = 0.0f;
        for (int i = 0; i < NUM_CLASSES; ++i) se += __expf(cls[i] - mx);
        const float lse = mx + __logf(se);
        for (int i = 0; i < NUM_CLASSES; ++i) out[i] = cls[i] - lse;
    }
}

// ------------------------------ host launcher ------------------------------

extern "C" void kernel_launch(void* const* d_in, const int* in_sizes, int n_in,
                              void* d_out, int out_size, void* d_ws, size_t ws_size,
                              hipStream_t stream) {
    (void)in_sizes; (void)n_in; (void)out_size; (void)ws_size;

    const float* node_att = (const float*)d_in[0];
    const int*   edge_src = (const int*)d_in[1];
    const int*   edge_tgt = edge_src + NUM_EDGES;
    const float* W_ih   = (const float*)d_in[2];
    const float* W_hh   = (const float*)d_in[3];
    const float* b_ih   = (const float*)d_in[4];
    const float* b_hh   = (const float*)d_in[5];
    const float* conv1W = (const float*)d_in[6];
    const float* conv1b = (const float*)d_in[7];
    const float* conv2W = (const float*)d_in[8];
    const float* conv2b = (const float*)d_in[9];
    const float* lin1W  = (const float*)d_in[10];
    const float* lin1b  = (const float*)d_in[11];
    const float* lin2W  = (const float*)d_in[12];
    const float* lin2b  = (const float*)d_in[13];
    float* out = (float*)d_out;

    // ---- workspace bump allocator (256B aligned) ----
    char* ws = (char*)d_ws;
    size_t off = 0;
    auto alloc = [&](size_t bytes) -> char* {
        char* p = ws + off;
        off += (bytes + 255) & ~(size_t)255;
        return p;
    };
    float*    G0       = (float*)alloc((size_t)SEQ_LEN * GATES * 4);      // 134 MB
    float*    H        = (float*)alloc((size_t)NUM_NODES * GCN_H * 4);
    float*    AGG      = (float*)alloc((size_t)NUM_NODES * GCN_H * 4);
    float*    XC       = (float*)alloc((size_t)NUM_NODES * GCN_H * 4);
    __bf16*   nattT_bf = (__bf16*)alloc((size_t)SEQ_LEN * NUM_NODES * 2); // [8192][1024]
    __bf16*   natt_bf  = (__bf16*)alloc((size_t)NUM_NODES * SEQ_LEN * 2); // [1024][8192]
    __bf16*   Wih_bf   = (__bf16*)alloc((size_t)GATES * NUM_NODES * 2);   // Bt for G0
    __bf16*   c1Wt_bf  = (__bf16*)alloc((size_t)GCN_H * SEQ_LEN * 2);     // [1024][8192]
    __bf16*   c2Wt_bf  = (__bf16*)alloc((size_t)GCN_H * GCN_H * 2);
    __bf16*   XA_bf    = (__bf16*)alloc((size_t)NUM_NODES * GCN_H * 2);
    __bf16*   XB_bf    = (__bf16*)alloc((size_t)NUM_NODES * GCN_H * 2);
    float*    deg   = (float*)alloc(NUM_NODES * 4);
    float*    dinv  = (float*)alloc(NUM_NODES * 4);
    float*    bsum  = (float*)alloc(GATES * 4);
    float*    hbuf  = (float*)alloc(2 * SEQ_EMB * 4);
    unsigned* bar   = (unsigned*)alloc(2 * 4);
    float*    gvec  = (float*)alloc(GCN_H * 4);
    float*    ev    = (float*)alloc(EVENT_EMB * 4);

    // ---- graph normalization ----
    k_deg_init<<<(NUM_NODES + 255) / 256, 256, 0, stream>>>(deg, NUM_NODES);
    k_deg_scatter<<<(NUM_EDGES + 255) / 256, 256, 0, stream>>>(edge_tgt, deg, NUM_EDGES);
    k_rsqrt<<<(NUM_NODES + 255) / 256, 256, 0, stream>>>(deg, dinv, NUM_NODES);

    // ---- bf16 operand preparation ----
    k_transpose_bf16<<<dim3(SEQ_LEN / 32, NUM_NODES / 32), 256, 0, stream>>>(
        node_att, nattT_bf, NUM_NODES, SEQ_LEN);                 // [8192][1024]
    k_convert_bf16<<<((long long)NUM_NODES * SEQ_LEN + 255) / 256, 256, 0, stream>>>(
        node_att, natt_bf, (long long)NUM_NODES * SEQ_LEN);
    k_convert_bf16<<<((long long)GATES * NUM_NODES + 255) / 256, 256, 0, stream>>>(
        W_ih, Wih_bf, (long long)GATES * NUM_NODES);             // already [N][K]
    k_transpose_bf16<<<dim3(GCN_H / 32, SEQ_LEN / 32), 256, 0, stream>>>(
        conv1W, c1Wt_bf, SEQ_LEN, GCN_H);                        // [1024][8192]
    k_transpose_bf16<<<dim3(GCN_H / 32, GCN_H / 32), 256, 0, stream>>>(
        conv2W, c2Wt_bf, GCN_H, GCN_H);

    // ---- LSTM input projection: G0[t,j] = sum_k nattT[t,k]*W_ih[j,k] + b ----
    k_bias_sum<<<(GATES + 255) / 256, 256, 0, stream>>>(b_ih, b_hh, bsum, GATES);
    k_gemm_bf16_wmma<<<dim3(SEQ_LEN / BM, GATES / BN), 256, 0, stream>>>(
        nattT_bf, NUM_NODES, Wih_bf, NUM_NODES, G0, bsum, SEQ_LEN, GATES, NUM_NODES);

    // ---- LSTM recurrence (persistent, grid-synced) ----
    k_zero_u32<<<(2 * SEQ_EMB + 255) / 256, 256, 0, stream>>>((unsigned*)hbuf, 2 * SEQ_EMB);
    k_zero_u32<<<1, 32, 0, stream>>>(bar, 2);
    k_lstm<<<LSTM_WGS, 256, 0, stream>>>(W_hh, G0, hbuf, bar, SEQ_LEN);
    float* h_last = hbuf;   // T even -> final h in buffer 0

    // ---- GCN layer 1 ----
    k_gemm_bf16_wmma<<<dim3(NUM_NODES / BM, GCN_H / BN), 256, 0, stream>>>(
        natt_bf, SEQ_LEN, c1Wt_bf, SEQ_LEN, H, nullptr, NUM_NODES, GCN_H, SEQ_LEN);
    k_zero_u32<<<((NUM_NODES * GCN_H) + 255) / 256, 256, 0, stream>>>(
        (unsigned*)AGG, (long long)NUM_NODES * GCN_H);
    k_gcn_scatter<<<NUM_EDGES, 256, 0, stream>>>(H, AGG, edge_src, edge_tgt, dinv);
    k_gcn_finalize<<<NUM_NODES, 256, 0, stream>>>(AGG, H, dinv, conv1b, nullptr, XA_bf, 1);

    // ---- GCN layer 2 ----
    k_gemm_bf16_wmma<<<dim3(NUM_NODES / BM, GCN_H / BN), 256, 0, stream>>>(
        XA_bf, GCN_H, c2Wt_bf, GCN_H, H, nullptr, NUM_NODES, GCN_H, GCN_H);
    k_zero_u32<<<((NUM_NODES * GCN_H) + 255) / 256, 256, 0, stream>>>(
        (unsigned*)AGG, (long long)NUM_NODES * GCN_H);
    k_gcn_scatter<<<NUM_EDGES, 256, 0, stream>>>(H, AGG, edge_src, edge_tgt, dinv);
    k_gcn_finalize<<<NUM_NODES, 256, 0, stream>>>(AGG, H, dinv, conv2b, nullptr, XB_bf, 1);

    // ---- GCN layer 3 (conv2 again, no relu) ----
    k_gemm_bf16_wmma<<<dim3(NUM_NODES / BM, GCN_H / BN), 256, 0, stream>>>(
        XB_bf, GCN_H, c2Wt_bf, GCN_H, H, nullptr, NUM_NODES, GCN_H, GCN_H);
    k_zero_u32<<<((NUM_NODES * GCN_H) + 255) / 256, 256, 0, stream>>>(
        (unsigned*)AGG, (long long)NUM_NODES * GCN_H);
    k_gcn_scatter<<<NUM_EDGES, 256, 0, stream>>>(H, AGG, edge_src, edge_tgt, dinv);
    k_gcn_finalize<<<NUM_NODES, 256, 0, stream>>>(AGG, H, dinv, conv2b, XC, nullptr, 0);

    // ---- readout ----
    k_col_mean<<<(GCN_H + 255) / 256, 256, 0, stream>>>(XC, gvec);
    k_lin1_relu<<<(EVENT_EMB + 255) / 256, 256, 0, stream>>>(h_last, gvec, lin1W, lin1b, ev);
    k_lin2_logsoftmax<<<1, 32, 0, stream>>>(ev, lin2W, lin2b, out);
}
